// HFOnlyAttn_79645873537311
// MI455X (gfx1250) — compile-verified, matrix operand order checked by reading, not run on hardware
//
#include <hip/hip_runtime.h>

typedef __attribute__((ext_vector_type(16))) _Float16 v16h;
typedef __attribute__((ext_vector_type(8)))  _Float16 v8h;
typedef __attribute__((ext_vector_type(8)))  float    v8f;
typedef __attribute__((ext_vector_type(4)))  float    f4;

#define CH   192
#define BD   64
#define HW   16384   // H*W = 128*128
#define WDIM 128

static __device__ __forceinline__ v8f wmma_f32d(v16h a, v16h b, v8f c) {
    // D(f32 16x16) = A(16x32 f16) * B(32x16 f16) + C
    return __builtin_amdgcn_wmma_f32_16x16x32_f16(false, a, false, b, (short)0, c, false, false);
}
static __device__ __forceinline__ v8h wmma_f16d(v16h a, v16h b, v8h c) {
    // D(f16 16x16) = A(16x32 f16) * B(32x16 f16) + C   -- packed 16-bit C/D layout
    return __builtin_amdgcn_wmma_f16_16x16x32_f16(false, a, false, b, (short)0, c, false, false);
}

static __device__ __forceinline__ v8h splat8(_Float16 x) {
    v8h r = {x, x, x, x, x, x, x, x};
    return r;
}

union V16 { v16h v; v8h h[2]; };

__global__ __launch_bounds__(256) void hf_attn_kernel(
    const float* __restrict__ hf,
    const float* __restrict__ Wqkv,
    const float* __restrict__ Wproj,
    const float* __restrict__ bproj,
    const float* __restrict__ resScale,
    float* __restrict__ out)
{
    extern __shared__ char smem[];
    _Float16* WtQ = (_Float16*)smem;                   // [192 cols][64 k] f16 (transposed Wqkv)
    _Float16* WtP = (_Float16*)(smem + 24576);         // [64 cols][64 k] f16 (transposed Wproj)
    float*    bP  = (float*)(smem + 24576 + 8192);     // [64]
    char* qkvArea = smem + 33024;                      // 8 waves * 18432 B

    const int tid    = threadIdx.x;
    const int lane   = tid & 31;
    const int waveId = tid >> 5;
    const int half   = lane >> 4;     // 0/1 : which half-wave
    const int lr     = lane & 15;     // tile row (A) / tile col (B)
    const int khalf  = half << 3;     // K sub-pattern offset: 0 or 8

    // ---- stage weights (block-wide, once) ----
    for (int i = tid; i < CH * BD; i += 256) {
        int k = i / CH, c = i - k * CH;                // Wqkv is (64 in, 192 out) row-major
        WtQ[c * BD + k] = (_Float16)Wqkv[i];
    }
    for (int i = tid; i < BD * BD; i += 256) {
        int k = i >> 6, c = i & 63;
        WtP[c * BD + k] = (_Float16)Wproj[i];
    }
    if (tid < BD) bP[tid] = bproj[tid];
    __syncthreads();

    const float rs = resScale[0];

    // wave-private LDS: 48 rows (3 bands x 16 pixels) x 192 f16, row stride 384 B.
    // bytes [0,128) of each row: q (later y, f16); [128,384): k|v (later proj result, f32)
    _Float16* qkv = (_Float16*)(qkvArea + waveId * (48 * CH * 2));

    const int pixBase = blockIdx.x * 128 + waveId * 16;   // 16 consecutive-w pixels
    const int bt  = pixBase >> 14;
    const int rem = pixBase & (HW - 1);
    const int hh  = rem >> 7;
    const int wb  = rem & (WDIM - 1);

    // ---------- Phase 1: load X (fp32 global, coalesced along w) -> f16 B registers ----------
    // B (K x 16) 16-bit layout: lane holds column n = pixel lr; elems 0..7 = K khalf..khalf+7,
    // elems 8..15 = K khalf+16..khalf+23 ; kc adds 32 to K.  (Mirrors the documented A pattern.)
    V16 Bx[3][2];
    const float* basePtr = hf + (size_t)bt * CH * HW + (size_t)hh * WDIM + wb + lr;
    #pragma unroll
    for (int band = 0; band < 3; ++band) {
        #pragma unroll
        for (int kc = 0; kc < 2; ++kc) {
            v16h a;
            #pragma unroll
            for (int i = 0; i < 8; ++i) {
                int c1 = (kc << 5) + khalf + i;
                a[i]     = (_Float16)basePtr[(size_t)(band * 64 + c1) * HW];
                a[i + 8] = (_Float16)basePtr[(size_t)(band * 64 + c1 + 16) * HW];
            }
            Bx[band][kc].v = a;
        }
    }

    // ---------- Phase 2: qkv GEMM  C = Wqkv^T (A) x X^T (B), f16 D ----------
    // 16-bit C/D layout: lane (col n = pixel lr) holds features m packed two-per-VGPR:
    // half=0 -> m 0..7, half=1 -> m 8..15  => v8h = 8 consecutive features, store b128 directly.
    for (int j = 0; j < 12; ++j) {
        const v8h* wcol = (const v8h*)(WtQ + (j * 16 + lr) * BD);  // A row m = out-feature j*16+lr
        V16 A0, A1;
        A0.h[0] = wcol[half];     A0.h[1] = wcol[half + 2];        // K 0..31 pattern
        A1.h[0] = wcol[half + 4]; A1.h[1] = wcol[half + 6];        // K 32..63 pattern
        #pragma unroll
        for (int band = 0; band < 3; ++band) {
            v8h acc = {(_Float16)0, (_Float16)0, (_Float16)0, (_Float16)0,
                       (_Float16)0, (_Float16)0, (_Float16)0, (_Float16)0};
            acc = wmma_f16d(A0.v, Bx[band][0].v, acc);
            acc = wmma_f16d(A1.v, Bx[band][1].v, acc);
            *(v8h*)(qkv + (band * 16 + lr) * CH + j * 16 + khalf) = acc;
        }
    }
    asm volatile("s_wait_dscnt 0x0" ::: "memory");

    // ---------- Phase 3: per-(pixel,head) 3x3 attention ----------
    #pragma unroll
    for (int uu = 0; uu < 2; ++uu) {
        const int id = lane + (uu << 5);          // 64 units = 16 pixels x 4 heads
        const int p  = id & 15;
        const int hd = id >> 4;
        v8h q0[3], q1[3], k0[3], k1[3], u0[3], u1[3];
        #pragma unroll
        for (int band = 0; band < 3; ++band) {
            const v8h* rowp = (const v8h*)(qkv + (band * 16 + p) * CH + hd * 16);
            q0[band] = rowp[0];  q1[band] = rowp[1];      // q: f 0..63
            k0[band] = rowp[8];  k1[band] = rowp[9];      // k: f 64..127
            u0[band] = rowp[16]; u1[band] = rowp[17];     // v: f 128..191
        }
        #pragma unroll
        for (int i = 0; i < 3; ++i) {
            float l0 = 0.f, l1 = 0.f, l2 = 0.f;
            #pragma unroll
            for (int d = 0; d < 8; ++d) {                 // f32 accumulate (v_fma_mix)
                l0 += (float)q0[i][d] * (float)k0[0][d] + (float)q1[i][d] * (float)k1[0][d];
                l1 += (float)q0[i][d] * (float)k0[1][d] + (float)q1[i][d] * (float)k1[1][d];
                l2 += (float)q0[i][d] * (float)k0[2][d] + (float)q1[i][d] * (float)k1[2][d];
            }
            l0 *= 0.25f; l1 *= 0.25f; l2 *= 0.25f;   // scale = 16^-0.5
            float m  = fmaxf(l0, fmaxf(l1, l2));
            float e0 = __expf(l0 - m), e1 = __expf(l1 - m), e2 = __expf(l2 - m);
            float inv = 1.f / (e0 + e1 + e2);
            // packed-f16 convex combination: y = e0*v0 + e1*v1 + e2*v2 (v_pk_fma_f16)
            v8h w0 = splat8((_Float16)(e0 * inv));
            v8h w1 = splat8((_Float16)(e1 * inv));
            v8h w2 = splat8((_Float16)(e2 * inv));
            v8h y0 = w0 * u0[0] + w1 * u0[1] + w2 * u0[2];
            v8h y1 = w0 * u1[0] + w1 * u1[1] + w2 * u1[2];
            v8h* dst = (v8h*)(qkv + (i * 16 + p) * CH + hd * 16);  // overwrite q in place
            dst[0] = y0; dst[1] = y1;
        }
    }
    asm volatile("s_wait_dscnt 0x0" ::: "memory");

    // ---------- Phase 4: proj GEMM  C = Wproj^T (A) x Y^T (B) -> f32 into dead k/v region ----------
    V16 By[3][2];                                        // Y B-tiles hoisted (lane = pixel)
    #pragma unroll
    for (int r = 0; r < 3; ++r) {
        const _Float16* rowbase = qkv + (r * 16 + lr) * CH;
        By[r][0].h[0] = *(const v8h*)(rowbase + khalf);
        By[r][0].h[1] = *(const v8h*)(rowbase + khalf + 16);
        By[r][1].h[0] = *(const v8h*)(rowbase + 32 + khalf);
        By[r][1].h[1] = *(const v8h*)(rowbase + 32 + khalf + 16);
    }
    #pragma unroll
    for (int j = 0; j < 4; ++j) {
        const v8h* wcol = (const v8h*)(WtP + (j * 16 + lr) * BD);
        V16 A0, A1;
        A0.h[0] = wcol[half];     A0.h[1] = wcol[half + 2];
        A1.h[0] = wcol[half + 4]; A1.h[1] = wcol[half + 6];
        #pragma unroll
        for (int r = 0; r < 3; ++r) {
            v8f acc = {0.f,0.f,0.f,0.f,0.f,0.f,0.f,0.f};
            acc = wmma_f32d(A0.v, By[r][0].v, acc);
            acc = wmma_f32d(A1.v, By[r][1].v, acc);
            // lane holds out-features j*16+8*half+0..7 for pixel lr; f32 row bytes [128,384)
            char* dst = (char*)qkv + (r * 16 + lr) * 384 + 128 + (j * 16 + khalf) * 4;
            f4 lo = {acc[0], acc[1], acc[2], acc[3]};
            f4 hi = {acc[4], acc[5], acc[6], acc[7]};
            *(f4*)dst       = lo;
            *((f4*)dst + 1) = hi;
        }
    }
    asm volatile("s_wait_dscnt 0x0" ::: "memory");

    // ---------- Phase 5: residual + bias, float4 transposed store (coalesced along W) ----------
    const size_t gbase = (size_t)bt * CH * HW + (size_t)hh * WDIM + wb;
    const int p4   = (lane & 3) << 2;   // 4 consecutive pixels per lane
    const int csub = lane >> 2;         // 8 channels per iteration across the wave
    for (int it = 0; it < 24; ++it) {
        int c = it * 8 + csub;
        int band = c >> 6;
        int f = c & 63;
        const char* yb = (const char*)qkv + (band * 16 + p4) * 384 + 128 + f * 4;
        float ya = *(const float*)yb;
        float y1 = *(const float*)(yb + 384);
        float y2 = *(const float*)(yb + 768);
        float y3 = *(const float*)(yb + 1152);
        float bias = bP[f];
        size_t g = gbase + (size_t)c * HW + p4;
        f4 hv = *(const f4*)(hf + g);
        f4 o;
        o.x = hv.x + rs * (ya + bias);
        o.y = hv.y + rs * (y1 + bias);
        o.z = hv.z + rs * (y2 + bias);
        o.w = hv.w + rs * (y3 + bias);
        *(f4*)(out + g) = o;
    }
}

extern "C" void kernel_launch(void* const* d_in, const int* in_sizes, int n_in,
                              void* d_out, int out_size, void* d_ws, size_t ws_size,
                              hipStream_t stream) {
    const float* hf    = (const float*)d_in[0];
    const float* Wqkv  = (const float*)d_in[1];
    const float* Wproj = (const float*)d_in[2];
    const float* bproj = (const float*)d_in[3];
    const float* rsc   = (const float*)d_in[4];
    float* out = (float*)d_out;

    const int nPix   = in_sizes[0] / CH;    // B*H*W = 131072
    const int blocks = nPix / 128;          // 128 pixels / block (8 waves x 16)
    const size_t smemBytes = 33024 + 8 * (48 * CH * 2);   // 180480 B (< 320 KB/WGP)

    hipFuncSetAttribute((const void*)hf_attn_kernel,
                        hipFuncAttributeMaxDynamicSharedMemorySize, (int)smemBytes);
    hf_attn_kernel<<<blocks, 256, smemBytes, stream>>>(hf, Wqkv, Wproj, bproj, rsc, out);
}
